// MultiHeadSelfAttention_49675591746258
// MI455X (gfx1250) — compile-verified, matrix operand order checked by reading, not run on hardware
//
#include <hip/hip_runtime.h>
#include <hip/hip_bf16.h>
#include <math.h>

typedef __attribute__((ext_vector_type(16))) __bf16 v16bf;
typedef __attribute__((ext_vector_type(8)))  __bf16 v8bf;
typedef __attribute__((ext_vector_type(8)))  float  v8f;
typedef __attribute__((ext_vector_type(4)))  float  v4f;

#define BB 2
#define SS 2048
#define EE 512
#define HH 8
#define DD 64

// ---- fragment loaders -------------------------------------------------------
// CDNA5 16-bit A/B fragment: lane L (n/m = L%16, g = L/16) holds 16 bf16:
//   elems 0..7  -> K = kbase + g*8 + i
//   elems 8..15 -> K = kbase + 16 + g*8 + (i-8)
// Each half is 16 contiguous bytes -> b128 loads.
static __device__ __forceinline__ v16bf frag_bf(const __bf16* row, int kbase, int g) {
  union { v16bf v; v8bf h[2]; } u;
  u.h[0] = *(const v8bf*)(row + kbase + g * 8);
  u.h[1] = *(const v8bf*)(row + kbase + 16 + g * 8);
  return u.v;
}

static __device__ __forceinline__ v16bf frag_f32(const float* row, int kbase, int g) {
  union { v16bf v; __bf16 e[16]; } u;
  v4f x0 = *(const v4f*)(row + kbase + g * 8);
  v4f x1 = *(const v4f*)(row + kbase + g * 8 + 4);
  v4f y0 = *(const v4f*)(row + kbase + 16 + g * 8);
  v4f y1 = *(const v4f*)(row + kbase + 16 + g * 8 + 4);
#pragma unroll
  for (int i = 0; i < 4; ++i) {
    u.e[i]      = (__bf16)x0[i];
    u.e[i + 4]  = (__bf16)x1[i];
    u.e[i + 8]  = (__bf16)y0[i];
    u.e[i + 12] = (__bf16)y1[i];
  }
  return u.v;
}

static __device__ __forceinline__ v8f wmma_bf16(v16bf a, v16bf b, v8f c) {
  return __builtin_amdgcn_wmma_f32_16x16x32_bf16(false, a, false, b, (short)0, c,
                                                 false, false);
}

// ---- stage 0: weight conversion fp32 -> bf16 --------------------------------
__global__ void cvt_f32_bf16(const float* __restrict__ src, __bf16* __restrict__ dst,
                             int n) {
  int i = blockIdx.x * blockDim.x + threadIdx.x;
  if (i < n) dst[i] = (__bf16)src[i];
}

// ---- stage 1: per-head projection Y = X(32768x64) @ Wv^T(64x64) -------------
// TR==0 -> out [B*H][S][D]; TR==1 -> out [B*H][D][S]  (compile-time: no branches)
template <int TR>
__global__ __launch_bounds__(256) void proj_kernel(const float* __restrict__ X,
                                                   const __bf16* __restrict__ Wv,
                                                   __bf16* __restrict__ out) {
  const int lane = threadIdx.x & 31;
  const int w    = threadIdx.x >> 5;
  const int ln   = lane & 15;
  const int g    = lane >> 4;
  const int m0   = (blockIdx.x * 8 + w) * 16;

  const float* arow = X + (size_t)(m0 + ln) * DD;
  v16bf a0 = frag_f32(arow, 0, g);
  v16bf a1 = frag_f32(arow, 32, g);

#pragma unroll
  for (int n = 0; n < 4; ++n) {
    const __bf16* brow = Wv + (size_t)(n * 16 + ln) * DD;  // B[d][e]=Wv[e][d]
    v16bf b0 = frag_bf(brow, 0, g);
    v16bf b1 = frag_bf(brow, 32, g);
    v8f c = {};
    c = wmma_bf16(a0, b0, c);
    c = wmma_bf16(a1, b1, c);
#pragma unroll
    for (int r = 0; r < 8; ++r) {
      int m  = m0 + r + 8 * g;                 // row = (b*S+s)*H + h
      int b  = m >> 14;                        // / (S*H)
      int s  = (m >> 3) & (SS - 1);
      int h  = m & (HH - 1);
      int bh = b * HH + h;
      __bf16 val = (__bf16)c[r];
      if (TR == 0)
        out[((size_t)bh * SS + s) * DD + n * 16 + ln] = val;
      else
        out[((size_t)bh * DD + n * 16 + ln) * SS + s] = val;
    }
  }
}

// ---- stage 2: flash attention per (b,h), 128-row q tile per block -----------
__global__ __launch_bounds__(256) void attn_kernel(const __bf16* __restrict__ qp,
                                                   const __bf16* __restrict__ kp,
                                                   const __bf16* __restrict__ vtp,
                                                   const int* __restrict__ mask,
                                                   __bf16* __restrict__ ao) {
  __shared__ float  sc[8][16][32];
  __shared__ __bf16 pbuf[8][16][32];
  __shared__ float  mrow[8][16];
  __shared__ float  lrow[8][16];
  __shared__ float  srow[8][16];
  __shared__ int    smask[SS];

  const int lane = threadIdx.x & 31;
  const int w    = threadIdx.x >> 5;
  const int ln   = lane & 15;
  const int g    = lane >> 4;
  const int bh   = blockIdx.x >> 4;  // 16 q-tiles of 128 rows
  const int qt   = blockIdx.x & 15;
  const int b    = bh >> 3;
  const int h    = bh & 7;

  for (int i = threadIdx.x; i < SS; i += 256) smask[i] = mask[b * SS + i];

  const __bf16* Q  = qp  + (size_t)bh * SS * DD;
  const __bf16* K  = kp  + (size_t)bh * SS * DD;
  const __bf16* VT = vtp + (size_t)bh * DD * SS;

  const int q0 = qt * 128 + w * 16;
  v16bf qa0 = frag_bf(Q + (size_t)(q0 + ln) * DD, 0, g);
  v16bf qa1 = frag_bf(Q + (size_t)(q0 + ln) * DD, 32, g);

  v8f o0 = {}, o1 = {}, o2 = {}, o3 = {};
  if (lane < 16) { mrow[w][lane] = -INFINITY; lrow[w][lane] = 0.0f; }
  __syncthreads();

  for (int kc = 0; kc < SS / 32; ++kc) {
    const int k0 = kc * 32;
    const int kn = (k0 + 32) & (SS - 1);  // next chunk (wraps; uniform, harmless)

    // prefetch next K rows and V^T row-segments while we compute this chunk
    __builtin_prefetch(K + (size_t)(kn + lane) * DD, 0, 3);
    __builtin_prefetch(VT + (size_t)(lane * 2) * SS + kn, 0, 3);
    __builtin_prefetch(VT + (size_t)(lane * 2 + 1) * SS + kn, 0, 3);

    // issue ALL K-fragment loads first so WMMAs can start on partial waits
    const __bf16* krow0 = K + (size_t)(k0 + ln) * DD;
    const __bf16* krow1 = K + (size_t)(k0 + 16 + ln) * DD;
    v16bf kb00 = frag_bf(krow0, 0, g);
    v16bf kb01 = frag_bf(krow0, 32, g);
    v16bf kb10 = frag_bf(krow1, 0, g);
    v16bf kb11 = frag_bf(krow1, 32, g);

    // scores: 16x32 tile = two 16x16 column groups, K=64 split in two
    v8f c0 = {}, c1 = {};
    c0 = wmma_bf16(qa0, kb00, c0);
    c0 = wmma_bf16(qa1, kb01, c0);
    c1 = wmma_bf16(qa0, kb10, c1);
    c1 = wmma_bf16(qa1, kb11, c1);
#pragma unroll
    for (int r = 0; r < 8; ++r) {
      sc[w][r + 8 * g][ln]      = c0[r];
      sc[w][r + 8 * g][16 + ln] = c1[r];
    }

    // V^T fragments don't depend on softmax: issue now, latency hides behind it
    v16bf vb0 = frag_bf(VT + (size_t)(0 * 16 + ln) * SS + k0, 0, g);
    v16bf vb1 = frag_bf(VT + (size_t)(1 * 16 + ln) * SS + k0, 0, g);
    v16bf vb2 = frag_bf(VT + (size_t)(2 * 16 + ln) * SS + k0, 0, g);
    v16bf vb3 = frag_bf(VT + (size_t)(3 * 16 + ln) * SS + k0, 0, g);
    __syncthreads();

    // online softmax update (one lane per row)
    if (lane < 16) {
      const int row = lane;
      float mold = mrow[w][row];
      float mx   = mold;
#pragma unroll
      for (int j = 0; j < 32; ++j) {
        float sv = smask[k0 + j] ? sc[w][row][j] : -INFINITY;
        mx = fmaxf(mx, sv);
      }
      float scale = (mx == -INFINITY) ? 1.0f : __expf(mold - mx);
      float lsum  = 0.0f;
#pragma unroll
      for (int j = 0; j < 32; ++j) {
        float pv = smask[k0 + j] ? __expf(sc[w][row][j] - mx) : 0.0f;
        lsum += pv;
        pbuf[w][row][j] = (__bf16)pv;
      }
      mrow[w][row] = mx;
      lrow[w][row] = lrow[w][row] * scale + lsum;
      srow[w][row] = scale;
    }
    __syncthreads();

    // rescale accumulators, then O += P(16x32) @ V(32x64)
    float fac[8];
#pragma unroll
    for (int r = 0; r < 8; ++r) fac[r] = srow[w][r + 8 * g];
#pragma unroll
    for (int r = 0; r < 8; ++r) {
      o0[r] *= fac[r]; o1[r] *= fac[r]; o2[r] *= fac[r]; o3[r] *= fac[r];
    }
    v16bf pa = frag_bf(&pbuf[w][ln][0], 0, g);
    o0 = wmma_bf16(pa, vb0, o0);
    o1 = wmma_bf16(pa, vb1, o1);
    o2 = wmma_bf16(pa, vb2, o2);
    o3 = wmma_bf16(pa, vb3, o3);
    __syncthreads();
  }

  float rcp[8];
#pragma unroll
  for (int r = 0; r < 8; ++r) {
    float l = lrow[w][r + 8 * g];
    rcp[r] = (l > 0.0f) ? 1.0f / l : 0.0f;
  }
#pragma unroll
  for (int r = 0; r < 8; ++r) {
    size_t base = ((size_t)(b * SS + q0 + r + 8 * g)) * EE + h * DD;
    ao[base + 0 * 16 + ln] = (__bf16)(o0[r] * rcp[r]);
    ao[base + 1 * 16 + ln] = (__bf16)(o1[r] * rcp[r]);
    ao[base + 2 * 16 + ln] = (__bf16)(o2[r] * rcp[r]);
    ao[base + 3 * 16 + ln] = (__bf16)(o3[r] * rcp[r]);
  }
}

// ---- stage 3: output projection (4096x512) @ Wo^T + bias, fp32 out ----------
__global__ __launch_bounds__(256) void outproj_kernel(const __bf16* __restrict__ ao,
                                                      const __bf16* __restrict__ wo,
                                                      const float* __restrict__ bo,
                                                      float* __restrict__ out) {
  const int lane = threadIdx.x & 31;
  const int w    = threadIdx.x >> 5;
  const int ln   = lane & 15;
  const int g    = lane >> 4;
  const int m0   = (blockIdx.x * 8 + w) * 16;

  const __bf16* arow = ao + (size_t)(m0 + ln) * EE;
  v16bf af[16];
#pragma unroll
  for (int kk = 0; kk < 16; ++kk) af[kk] = frag_bf(arow, kk * 32, g);

  for (int n = 0; n < 32; ++n) {
    float bias = bo[n * 16 + ln];
    v8f c;
#pragma unroll
    for (int r = 0; r < 8; ++r) c[r] = bias;
    const __bf16* brow = wo + (size_t)(n * 16 + ln) * EE;  // B[j][e]=Wo[e][j]
#pragma unroll
    for (int kk = 0; kk < 16; ++kk) {
      v16bf bf = frag_bf(brow, kk * 32, g);
      c = wmma_bf16(af[kk], bf, c);
    }
#pragma unroll
    for (int r = 0; r < 8; ++r)
      out[(size_t)(m0 + r + 8 * g) * EE + n * 16 + ln] = c[r];
  }
}

// ---- host side --------------------------------------------------------------
extern "C" void kernel_launch(void* const* d_in, const int* in_sizes, int n_in,
                              void* d_out, int out_size, void* d_ws, size_t ws_size,
                              hipStream_t stream) {
  const float* values  = (const float*)d_in[0];
  const float* keys    = (const float*)d_in[1];
  const float* queries = (const float*)d_in[2];
  const int*   mask    = (const int*)d_in[3];
  const float* Wv      = (const float*)d_in[4];
  const float* Wo      = (const float*)d_in[5];
  const float* bo      = (const float*)d_in[6];
  float* out = (float*)d_out;

  // workspace layout (bf16 elements, all 16B aligned)
  __bf16* ws   = (__bf16*)d_ws;
  __bf16* wv_b = ws;                 //   4096
  __bf16* wo_b = ws + 4096;          // 262144
  __bf16* qp   = ws + 266240;        // 2097152  [B*H][S][D]
  __bf16* kp   = ws + 2363392;       // 2097152  [B*H][S][D]
  __bf16* vtp  = ws + 4460544;       // 2097152  [B*H][D][S]
  __bf16* ao   = ws + 6557696;       // 2097152  [B*S][E]

  cvt_f32_bf16<<<(4096 + 255) / 256, 256, 0, stream>>>(Wv, wv_b, 4096);
  cvt_f32_bf16<<<(262144 + 255) / 256, 256, 0, stream>>>(Wo, wo_b, 262144);

  // 32768 rows / 16 per wave / 8 waves per block = 256 blocks
  proj_kernel<0><<<256, 256, 0, stream>>>(queries, wv_b, qp);
  proj_kernel<0><<<256, 256, 0, stream>>>(keys,    wv_b, kp);
  proj_kernel<1><<<256, 256, 0, stream>>>(values,  wv_b, vtp);

  // 16 (b,h) pairs * 16 q-tiles of 128 rows = 256 blocks
  attn_kernel<<<256, 256, 0, stream>>>(qp, kp, vtp, mask, ao);

  // 4096 rows / 128 per block = 32 blocks
  outproj_kernel<<<32, 256, 0, stream>>>(ao, wo_b, bo, out);
}
